// CrossAttentionFusion_11020886081869
// MI455X (gfx1250) — matrix operand order repaired and checked by reading, not compiled, not run on hardware
//
#include <hip/hip_runtime.h>
#include <math.h>

// ---------------------------------------------------------------------------
// CrossAttentionFusion for MI455X (gfx1250, wave32, WMMA).
//
// seq_len==1 MHA => softmax == 1.0 exactly => attn outputs are constant 1.0
// and each attention block collapses to (x @ wv^T + bv) @ wo^T + bo.
//
// Roofline: 81.6 GFLOP vs ~175 MB traffic => compute bound => bf16 WMMA
// (v_wmma_f32_16x16x32_bf16, f32 accum).  Weights (~5 MB bf16) live in L2.
//
//  * A stripe [16 x K] staged into LDS via GLOBAL_LOAD_ASYNC_TO_LDS_B128
//    (ASYNCcnt + s_wait_asynccnt), deduping the 8x redundant A loads.
//  * B (weight) fragments double-buffered in registers so the next k-slice's
//    loads are in flight during the current WMMAs.
//  * LDS A-buffer aliased with the epilogue f32 tile (barrier between phases).
// ---------------------------------------------------------------------------

typedef __bf16 bf16_t;
typedef __attribute__((ext_vector_type(16))) __bf16 v16bf;
typedef __attribute__((ext_vector_type(8)))  float  v8f;
typedef __attribute__((ext_vector_type(4)))  unsigned int v4u;
typedef int v4i __attribute__((vector_size(16)));   // matches builtin param type

#define AS1 __attribute__((address_space(1)))
#define AS3 __attribute__((address_space(3)))

#if defined(__has_builtin)
#if __has_builtin(__builtin_amdgcn_global_load_async_to_lds_b128)
#define HAVE_ASYNC_LDS 1
#endif
#endif
#ifndef HAVE_ASYNC_LDS
#define HAVE_ASYNC_LDS 0
#endif

#define BATCH 16384
#define TDIM  256
#define FDIM  512
#define H2DIM 256

__device__ __forceinline__ unsigned short f2bf(float f) {
  union { float f; unsigned u; } x; x.f = f;
  unsigned r = x.u + 0x7FFFu + ((x.u >> 16) & 1u);   // round to nearest even
  return (unsigned short)(r >> 16);
}
__device__ __forceinline__ float bf2f(unsigned short h) {
  union { unsigned u; float f; } x; x.u = ((unsigned)h) << 16;
  return x.f;
}

__device__ __forceinline__ void wait_async0() {
#if __has_builtin(__builtin_amdgcn_s_wait_asynccnt)
  __builtin_amdgcn_s_wait_asynccnt(0);
#else
  asm volatile("s_wait_asynccnt 0x0" ::: "memory");
#endif
}

// A fragment (16x32 bf16, M x K) from LDS: lane = M row; lanes 0-15 hold
// K=[k..k+7],[k+16..k+23]; lanes 16-31 hold K=[k+8..k+15],[k+24..k+31].
__device__ __forceinline__ v16bf load_frag_a_lds(const unsigned short* sA,
                                                 int lda, int k, int lane) {
  const int lrow = lane & 15, lh = lane >> 4;
  union { v4u u[2]; v16bf v; } f;
  const unsigned short* p = sA + lrow * lda + k + lh * 8;
  f.u[0] = *(const v4u*)p;
  f.u[1] = *(const v4u*)(p + 16);
  return f.v;
}

// B fragment (32x16 bf16, K x N): lane%16 = N col; lanes 0-15 hold K=[k..k+7],[k+16..k+23],
// lanes 16-31 hold K=[k+8..k+15],[k+24..k+31]. W stored [N][K] row-major.
__device__ __forceinline__ v16bf load_frag_b(const unsigned short* W, int ldw,
                                             int n0, int k, int lane) {
  const int lrow = lane & 15, lh = lane >> 4;
  union { v4u u[2]; v16bf v; } f;
  const unsigned short* p = W + (size_t)(n0 + lrow) * ldw + k + lh * 8;
  f.u[0] = *(const v4u*)p;
  f.u[1] = *(const v4u*)(p + 16);
  return f.v;
}

// ---------------------------------------------------------------------------
// Fused GEMM: OUT[16 x N] = act( maybeLN( A[16 x K] @ W^T + bias + maybeRes ) )
// Block = 256 threads = 8 waves; block owns full N so LayerNorm row-reductions
// stay in-block.  Wave w covers columns [w*N/8, (w+1)*N/8).
// ACT: 0=none, 1=sigmoid, 2=exact gelu.
// ---------------------------------------------------------------------------
template <int K, int N, bool SPLIT_A, bool HAS_RES, bool DO_LN, int ACT, bool OUT_F32>
__global__ __launch_bounds__(256) void fused_gemm(
    const unsigned short* __restrict__ A0,   // bf16 [B x K] (or [B x K/2] if split)
    const unsigned short* __restrict__ A1,   // second half of concat (or null)
    const unsigned short* __restrict__ W,    // bf16 [N x K] row-major
    const float* __restrict__ bias,          // [N]
    const unsigned short* __restrict__ Res,  // bf16 [B x N] residual (or null)
    const float* __restrict__ ln_g, const float* __restrict__ ln_b,
    unsigned short* __restrict__ outB, float* __restrict__ outF) {
  constexpr int TN  = N / 128;               // n-tiles (of 16) per wave
  constexpr int LDA = K + 8;                 // padded LDS row stride (avoid bank dup)
  constexpr unsigned A_BYTES = 16u * LDA * 2u;
  constexpr unsigned T_BYTES = 16u * N * 4u;
  constexpr unsigned SMEM_BYTES = (A_BYTES > T_BYTES) ? A_BYTES : T_BYTES;
  __shared__ __align__(16) unsigned char smem[SMEM_BYTES];   // sA, then s_tile
  __shared__ float s_part[2][16 * 16];
  __shared__ float s_mr[2][16];
  unsigned short* sA     = (unsigned short*)smem;
  float*          s_tile = (float*)smem;

  const int tid  = threadIdx.x;
  const int wave = tid >> 5;
  const int lane = tid & 31;
  const int row0 = blockIdx.x * 16;
  const int lrow = lane & 15, lh = lane >> 4;

  // --- stage the A stripe [16 x K] into LDS (async copy, 16B chunks) -------
  constexpr int CHUNKS = 16 * (K / 8);       // 16B chunks
  for (int c = tid; c < CHUNKS; c += 256) {
    const int row = c / (K / 8);
    const int j   = (c - row * (K / 8)) * 8;   // element offset within row
    const unsigned short* gsrc;
    if (SPLIT_A && j >= K / 2)
      gsrc = A1 + (size_t)(row0 + row) * (K / 2) + (j - K / 2);
    else
      gsrc = A0 + (size_t)(row0 + row) * (SPLIT_A ? (K / 2) : K) + j;
    unsigned short* ldst = sA + row * LDA + j;
#if HAVE_ASYNC_LDS
    __builtin_amdgcn_global_load_async_to_lds_b128(
        (AS1 v4i*)gsrc, (AS3 v4i*)ldst, 0, 0);
#else
    *(v4u*)ldst = *(const v4u*)gsrc;
#endif
  }
#if HAVE_ASYNC_LDS
  wait_async0();
#endif
  __syncthreads();

  // --- K loop: A from LDS, B double-buffered from global (L2-resident) -----
  v8f acc[TN];
#pragma unroll
  for (int i = 0; i < TN; ++i) acc[i] = (v8f){0.f,0.f,0.f,0.f,0.f,0.f,0.f,0.f};

  v16bf bcur[TN];
#pragma unroll
  for (int i = 0; i < TN; ++i)
    bcur[i] = load_frag_b(W, K, (wave * TN + i) * 16, 0, lane);

#pragma unroll
  for (int k = 0; k < K; k += 32) {
    v16bf bnxt[TN];
    if (k + 32 < K) {
#pragma unroll
      for (int i = 0; i < TN; ++i)
        bnxt[i] = load_frag_b(W, K, (wave * TN + i) * 16, k + 32, lane);
    }
    if (k + 64 < K)  // prefetch k+64 weight slice (global_prefetch_b8)
      __builtin_prefetch(W + (size_t)(wave * TN * 16 + lrow) * K + (k + 64) + lh * 16, 0, 1);
    const v16bf a = load_frag_a_lds(sA, LDA, k, lane);
#pragma unroll
    for (int i = 0; i < TN; ++i)
      acc[i] = __builtin_amdgcn_wmma_f32_16x16x32_bf16(
          false, a, false, bcur[i], (short)0, acc[i], false, false);
    if (k + 32 < K) {
#pragma unroll
      for (int i = 0; i < TN; ++i) bcur[i] = bnxt[i];
    }
  }
  __syncthreads();   // sA dead beyond this point; smem becomes s_tile

  // --- accumulators (+bias) -> LDS f32 tile.  C layout: lane%16 = col,
  // VGPR r -> row r + 8*(lane/16)  (ISA 7.12.2).
#pragma unroll
  for (int i = 0; i < TN; ++i) {
    const int col = (wave * TN + i) * 16 + lrow;
    const float bv = bias[col];
#pragma unroll
    for (int r = 0; r < 8; ++r) {
      const int row = r + 8 * lh;
      s_tile[row * N + col] = acc[i][r] + bv;
    }
  }
  __syncthreads();

  if constexpr (HAS_RES) {   // residual added BEFORE LayerNorm (coalesced)
    for (int idx = tid; idx < 16 * N; idx += 256) {
      const int row = idx / N, col = idx - row * N;
      s_tile[idx] += bf2f(Res[(size_t)(row0 + row) * N + col]);
    }
    __syncthreads();
  }

  if constexpr (DO_LN) {
    const int row = tid >> 4, c0 = tid & 15;
    float s = 0.f, q = 0.f;
    for (int col = c0; col < N; col += 16) {
      const float x = s_tile[row * N + col];
      s += x; q += x * x;
    }
    s_part[0][tid] = s; s_part[1][tid] = q;
    __syncthreads();
    if (tid < 16) {
      float ss = 0.f, qq = 0.f;
      for (int j = 0; j < 16; ++j) { ss += s_part[0][tid * 16 + j]; qq += s_part[1][tid * 16 + j]; }
      const float mean = ss / (float)N;
      const float var  = qq / (float)N - mean * mean;
      s_mr[0][tid] = mean;
      s_mr[1][tid] = rsqrtf(var + 1e-5f);
    }
    __syncthreads();
  }

  for (int idx = tid; idx < 16 * N; idx += 256) {
    const int row = idx / N, col = idx - row * N;
    float x = s_tile[idx];
    if constexpr (DO_LN)
      x = (x - s_mr[0][row]) * s_mr[1][row] * ln_g[col] + ln_b[col];
    if constexpr (ACT == 1) x = 1.0f / (1.0f + __expf(-x));
    else if constexpr (ACT == 2) x = 0.5f * x * (1.0f + erff(x * 0.70710678118654752f));
    const size_t o = (size_t)(row0 + row) * N + col;
    if constexpr (OUT_F32) outF[o] = x;
    else                   outB[o] = f2bf(x);
  }
}

__global__ void convert_f32_bf16(const float* __restrict__ in,
                                 unsigned short* __restrict__ out, int n) {
  const int i = blockIdx.x * blockDim.x + threadIdx.x;
  if (i < n) out[i] = f2bf(in[i]);
}

__global__ void fill_ones(float* __restrict__ p, int n) {
  const int i = blockIdx.x * blockDim.x + threadIdx.x;
  if (i < n) p[i] = 1.0f;
}

// ---------------------------------------------------------------------------
extern "C" void kernel_launch(void* const* d_in, const int* in_sizes, int n_in,
                              void* d_out, int out_size, void* d_ws, size_t ws_size,
                              hipStream_t stream) {
  (void)in_sizes; (void)n_in; (void)out_size; (void)ws_size;
  // input order: 0 text_emb, 1 num_emb, 2 tp_w, 3 tp_b, 4 np_w, 5 np_b,
  // 6 a1_wq, 7 a1_wk, 8 a1_wv, 9 a1_bq, 10 a1_bk, 11 a1_bv, 12 a1_wo,
  // 13 a1_bo, 14 a2_wq, 15 a2_wk, 16 a2_wv, 17 a2_bq, 18 a2_bk, 19 a2_bv,
  // 20 a2_wo, 21 a2_bo, 22 n1_g, 23 n1_b, 24 n2_g, 25 n2_b, 26 g_w, 27 g_b,
  // 28 m1_w, 29 m1_b, 30 ln1_g, 31 ln1_b, 32 m2_w, 33 m2_b, 34 ln2_g, 35 ln2_b
  const float* f_text = (const float*)d_in[0];
  const float* f_num  = (const float*)d_in[1];
  const float* f_tpw  = (const float*)d_in[2];
  const float* f_tpb  = (const float*)d_in[3];
  const float* f_npw  = (const float*)d_in[4];
  const float* f_npb  = (const float*)d_in[5];
  const float* f_a1wv = (const float*)d_in[8];
  const float* f_a1bv = (const float*)d_in[11];
  const float* f_a1wo = (const float*)d_in[12];
  const float* f_a1bo = (const float*)d_in[13];
  const float* f_a2wv = (const float*)d_in[16];
  const float* f_a2bv = (const float*)d_in[19];
  const float* f_a2wo = (const float*)d_in[20];
  const float* f_a2bo = (const float*)d_in[21];
  const float* f_n1g  = (const float*)d_in[22];
  const float* f_n1b  = (const float*)d_in[23];
  const float* f_n2g  = (const float*)d_in[24];
  const float* f_n2b  = (const float*)d_in[25];
  const float* f_gw   = (const float*)d_in[26];
  const float* f_gb   = (const float*)d_in[27];
  const float* f_m1w  = (const float*)d_in[28];
  const float* f_m1b  = (const float*)d_in[29];
  const float* f_ln1g = (const float*)d_in[30];
  const float* f_ln1b = (const float*)d_in[31];
  const float* f_m2w  = (const float*)d_in[32];
  const float* f_m2b  = (const float*)d_in[33];
  const float* f_ln2g = (const float*)d_in[34];
  const float* f_ln2b = (const float*)d_in[35];

  // output: fused [B,256] | gate [B,512] | attn_n2t [B] | attn_t2n [B]
  float* out_fused = (float*)d_out;
  float* out_gate  = out_fused + (size_t)BATCH * H2DIM;
  float* out_attn  = out_gate + (size_t)BATCH * FDIM;   // 2*B ones

  // workspace layout (bf16 / ushort elements)
  unsigned short* ws = (unsigned short*)d_ws;
  const size_t off_text = 0;
  const size_t off_num  = off_text + (size_t)BATCH * TDIM;
  const size_t off_tp   = off_num  + (size_t)BATCH * TDIM;       // weights...
  const size_t off_np   = off_tp   + (size_t)FDIM * TDIM;
  const size_t off_a1wv = off_np   + (size_t)FDIM * TDIM;
  const size_t off_a1wo = off_a1wv + (size_t)FDIM * FDIM;
  const size_t off_a2wv = off_a1wo + (size_t)FDIM * FDIM;
  const size_t off_a2wo = off_a2wv + (size_t)FDIM * FDIM;
  const size_t off_gw   = off_a2wo + (size_t)FDIM * FDIM;
  const size_t off_m1w  = off_gw   + (size_t)FDIM * 2 * FDIM;
  const size_t off_m2w  = off_m1w  + (size_t)FDIM * 2 * FDIM;
  const size_t off_t    = off_m2w  + (size_t)H2DIM * FDIM;       // activations
  const size_t off_n    = off_t    + (size_t)BATCH * FDIM;
  const size_t off_n2t  = off_n    + (size_t)BATCH * FDIM;
  const size_t off_t2n  = off_n2t  + (size_t)BATCH * FDIM;
  const size_t off_v    = off_text;  // reuse emb region once t,n are built
  const size_t off_h    = off_text;  // reuse again once v is dead

  auto cvt = [&](const float* src, size_t dstOff, int n) {
    convert_f32_bf16<<<(n + 255) / 256, 256, 0, stream>>>(src, ws + dstOff, n);
  };
  cvt(f_text, off_text, BATCH * TDIM);
  cvt(f_num,  off_num,  BATCH * TDIM);
  cvt(f_tpw,  off_tp,   FDIM * TDIM);
  cvt(f_npw,  off_np,   FDIM * TDIM);
  cvt(f_a1wv, off_a1wv, FDIM * FDIM);
  cvt(f_a1wo, off_a1wo, FDIM * FDIM);
  cvt(f_a2wv, off_a2wv, FDIM * FDIM);
  cvt(f_a2wo, off_a2wo, FDIM * FDIM);
  cvt(f_gw,   off_gw,   FDIM * 2 * FDIM);
  cvt(f_m1w,  off_m1w,  FDIM * 2 * FDIM);
  cvt(f_m2w,  off_m2w,  H2DIM * FDIM);

  fill_ones<<<(2 * BATCH + 255) / 256, 256, 0, stream>>>(out_attn, 2 * BATCH);

  const dim3 grid(BATCH / 16), blk(256);

  // t = text @ tp^T + b ; n = num @ np^T + b
  fused_gemm<TDIM, FDIM, false, false, false, 0, false><<<grid, blk, 0, stream>>>(
      ws + off_text, nullptr, ws + off_tp, f_tpb, nullptr, nullptr, nullptr,
      ws + off_t, nullptr);
  fused_gemm<TDIM, FDIM, false, false, false, 0, false><<<grid, blk, 0, stream>>>(
      ws + off_num, nullptr, ws + off_np, f_npb, nullptr, nullptr, nullptr,
      ws + off_n, nullptr);

  // v1 = t @ a1_wv^T + b ;  n2t = LN(n + v1 @ a1_wo^T + b)
  fused_gemm<FDIM, FDIM, false, false, false, 0, false><<<grid, blk, 0, stream>>>(
      ws + off_t, nullptr, ws + off_a1wv, f_a1bv, nullptr, nullptr, nullptr,
      ws + off_v, nullptr);
  fused_gemm<FDIM, FDIM, false, true, true, 0, false><<<grid, blk, 0, stream>>>(
      ws + off_v, nullptr, ws + off_a1wo, f_a1bo, ws + off_n, f_n1g, f_n1b,
      ws + off_n2t, nullptr);

  // v2 = n @ a2_wv^T + b ;  t2n = LN(t + v2 @ a2_wo^T + b)
  fused_gemm<FDIM, FDIM, false, false, false, 0, false><<<grid, blk, 0, stream>>>(
      ws + off_n, nullptr, ws + off_a2wv, f_a2bv, nullptr, nullptr, nullptr,
      ws + off_v, nullptr);
  fused_gemm<FDIM, FDIM, false, true, true, 0, false><<<grid, blk, 0, stream>>>(
      ws + off_v, nullptr, ws + off_a2wo, f_a2bo, ws + off_t, f_n2g, f_n2b,
      ws + off_t2n, nullptr);

  // gate = sigmoid([n2t|t2n] @ g_w^T + b)   -> f32 out
  fused_gemm<2 * FDIM, FDIM, true, false, false, 1, true><<<grid, blk, 0, stream>>>(
      ws + off_n2t, ws + off_t2n, ws + off_gw, f_gb, nullptr, nullptr, nullptr,
      nullptr, out_gate);

  // h = gelu(LN([n2t|t2n] @ m1^T + b))
  fused_gemm<2 * FDIM, FDIM, true, false, true, 2, false><<<grid, blk, 0, stream>>>(
      ws + off_n2t, ws + off_t2n, ws + off_m1w, f_m1b, nullptr, f_ln1g, f_ln1b,
      ws + off_h, nullptr);

  // fused = gelu(LN(h @ m2^T + b))          -> f32 out
  fused_gemm<FDIM, H2DIM, false, false, true, 2, true><<<grid, blk, 0, stream>>>(
      ws + off_h, nullptr, ws + off_m2w, f_m2b, nullptr, f_ln2g, f_ln2b,
      nullptr, out_fused);
}
